// Scanner_encoder_72885595013191
// MI455X (gfx1250) — matrix operand order repaired and checked by reading, not compiled, not run on hardware
//
#include <hip/hip_runtime.h>
#include <hip/hip_bf16.h>

typedef __attribute__((ext_vector_type(16))) __bf16 v16bf;
typedef __attribute__((ext_vector_type(8)))  float  v8f;

// ---------------------------------------------------------------------------
// Fused gather(x[no]) -> GEMM (y = mat @ w^T + b) using v_wmma_f32_16x16x32_bf16
// One wave computes one 16x16 output tile. K = 7*cin, cin is a power of two.
// Fast path (cin >= 32): neighbor-slot j is hoisted to an unrolled outer loop,
// so the inner K loop does only aligned float4 loads + cvt + wmma.
// ---------------------------------------------------------------------------
__global__ void __launch_bounds__(256)
gemm_gather_wmma(const float* __restrict__ x,    // (Nv, cin)
                 const int*   __restrict__ no,   // (7*Nv)
                 const float* __restrict__ w,    // (cout, 7*cin) row-major
                 const float* __restrict__ bias, // (cout)
                 float*       __restrict__ y,    // (Nv, cout)
                 int Nv, int cin, int lc, int cout)
{
    const int K    = 7 * cin;
    const int lane = threadIdx.x & 31;
    const int wave = threadIdx.x >> 5;
    const int wpb  = blockDim.x >> 5;
    const int tilesM = (Nv + 15) >> 4;
    const int tilesN = cout >> 4;
    const long long totalTiles = (long long)tilesM * tilesN;
    const long long tileId = (long long)blockIdx.x * wpb + wave;  // wave-uniform
    if (tileId >= totalTiles) return;                             // uniform exit
    const int tM   = (int)(tileId % tilesM);
    const int tN   = (int)(tileId / tilesM);
    const int lrow = tM * 16 + (lane & 15);
    const int col  = tN * 16 + (lane & 15);
    const int half = lane >> 4;
    const int rclamp = (lrow < Nv) ? lrow : (Nv - 1);
    const int* nrow = no + 7 * rclamp;

    // B rows are the K axis of w (B = w^T); this lane's half owns K-offsets
    // {half*8+0..7, 16+half*8+0..7} of each 32-wide K step.
    const float* wrow = w + (long long)col * K + half * 8;

    v8f acc = {};
    if (cin >= 32) {
        // Hoist all 7 neighbor indices into registers (j-loop fully unrolled).
        int nidx[7];
        #pragma unroll
        for (int j = 0; j < 7; ++j) nidx[j] = nrow[j];

        #pragma unroll
        for (int j = 0; j < 7; ++j) {
            // A-fragment ISA layout (16-bit 16x32): this lane's half owns
            // K-offsets {half*8+0..7, 16+half*8+0..7} of each 32-wide step.
            const float* arow = x + (long long)nidx[j] * cin + half * 8;
            const float* wj   = wrow + j * cin;
            #pragma unroll 2
            for (int c0 = 0; c0 < cin; c0 += 32) {
                v16bf a, b;
                const float4* p0 = (const float4*)(arow + c0);       // K run 0
                const float4* p1 = (const float4*)(arow + c0 + 16);  // K run 1
                const float4 f0 = p0[0], f1 = p0[1];
                const float4 f2 = p1[0], f3 = p1[1];
                a[0]  = (__bf16)f0.x; a[1]  = (__bf16)f0.y;
                a[2]  = (__bf16)f0.z; a[3]  = (__bf16)f0.w;
                a[4]  = (__bf16)f1.x; a[5]  = (__bf16)f1.y;
                a[6]  = (__bf16)f1.z; a[7]  = (__bf16)f1.w;
                a[8]  = (__bf16)f2.x; a[9]  = (__bf16)f2.y;
                a[10] = (__bf16)f2.z; a[11] = (__bf16)f2.w;
                a[12] = (__bf16)f3.x; a[13] = (__bf16)f3.y;
                a[14] = (__bf16)f3.z; a[15] = (__bf16)f3.w;

                const float4* q = (const float4*)(wj + c0);
                #pragma unroll
                for (int r = 0; r < 4; ++r) {
                    const float4 f = q[(r < 2) ? r : (r + 2)];
                    b[r*4+0] = (__bf16)f.x; b[r*4+1] = (__bf16)f.y;
                    b[r*4+2] = (__bf16)f.z; b[r*4+3] = (__bf16)f.w;
                }
                acc = __builtin_amdgcn_wmma_f32_16x16x32_bf16(
                    false, a, false, b, (short)0, acc, false, false);
            }
        }
    } else {
        // Level-0 convA only: cin=2, K=14 -> single masked K-step.
        // Branch-free: clamp index so loads are always valid, then select.
        for (int k0 = 0; k0 < K; k0 += 32) {
            v16bf a, b;
            #pragma unroll
            for (int e = 0; e < 16; ++e) {
                const int k  = k0 + ((e < 8) ? (half * 8 + e) : (16 + half * 8 + (e - 8)));
                const int kc = (k < K) ? k : 0;
                const int j  = kc >> lc, c = kc & (cin - 1);
                const float v = x[(long long)nrow[j] * cin + c];
                a[e] = (__bf16)((k < K) ? v : 0.0f);
            }
            #pragma unroll
            for (int e = 0; e < 16; ++e) {
                const int k  = k0 + ((e < 8) ? (half * 8 + e) : (16 + half * 8 + (e - 8)));
                const int kc = (k < K) ? k : 0;
                const float v = w[(long long)col * K + kc];
                b[e] = (__bf16)((k < K) ? v : 0.0f);
            }
            acc = __builtin_amdgcn_wmma_f32_16x16x32_bf16(
                false, a, false, b, (short)0, acc, false, false);
        }
    }

    // C/D layout: lane l, VGPR v -> row = v + 8*half, col = l&15.
    const float bv = bias[col];
    #pragma unroll
    for (int v = 0; v < 8; ++v) {
        const int m = tM * 16 + v + 8 * half;
        if (m < Nv) y[(long long)m * cout + col] = acc[v] + bv;
    }
}

// ---------------------------------------------------------------------------
// GroupNorm(4 groups) + LeakyReLU(0.2), in place. One wave per vertex.
// ---------------------------------------------------------------------------
__global__ void __launch_bounds__(256)
gn_leaky(float* __restrict__ y, const float* __restrict__ gamma,
         const float* __restrict__ beta, int Nv, int cout)
{
    const int lane = threadIdx.x & 31;
    const int wave = threadIdx.x >> 5;
    const int wpb  = blockDim.x >> 5;
    const int vtx  = blockIdx.x * wpb + wave;
    if (vtx >= Nv) return;
    float* row = y + (long long)vtx * cout;
    const int gsz = cout >> 2;  // channels per group
    for (int g = 0; g < 4; ++g) {
        float s = 0.f, ss = 0.f;
        for (int c = lane; c < gsz; c += 32) {
            const float v = row[g * gsz + c];
            s += v; ss += v * v;
        }
        #pragma unroll
        for (int off = 16; off >= 1; off >>= 1) {
            s  += __shfl_xor(s,  off, 32);
            ss += __shfl_xor(ss, off, 32);
        }
        const float mu  = s / (float)gsz;
        const float var = ss / (float)gsz - mu * mu;  // population variance (matches xg.var)
        const float inv = rsqrtf(var + 1e-5f);
        for (int c = lane; c < gsz; c += 32) {
            const int ch = g * gsz + c;
            float v = (row[ch] - mu) * inv * gamma[ch] + beta[ch];
            row[ch] = (v >= 0.f) ? v : 0.2f * v;
        }
    }
}

// ---------------------------------------------------------------------------
// pool_mean: out[i,c] = mean_j x[no_fine[7i+j], c]
// ---------------------------------------------------------------------------
__global__ void __launch_bounds__(256)
pool_mean_kernel(const float* __restrict__ x, const int* __restrict__ no,
                 float* __restrict__ out, int n_coarse, int C)
{
    const long long idx = (long long)blockIdx.x * blockDim.x + threadIdx.x;
    const long long total = (long long)n_coarse * C;
    if (idx >= total) return;
    const int i = (int)(idx / C);
    const int c = (int)(idx - (long long)i * C);
    float s = 0.f;
    #pragma unroll
    for (int j = 0; j < 7; ++j) s += x[(long long)no[7 * i + j] * C + c];
    out[idx] = s * (1.0f / 7.0f);
}

// ---------------------------------------------------------------------------
// Head: latent = mean over channels, out = sigmoid(latent @ wout^T + bout)
// d_out = [out (S) | latent (Nv)]
// ---------------------------------------------------------------------------
__global__ void __launch_bounds__(256)
head_kernel(const float* __restrict__ x, const float* __restrict__ wout,
            const float* __restrict__ bout, float* __restrict__ dout,
            int Nv, int C, int S)
{
    __shared__ float lat[64];
    const int lane = threadIdx.x & 31;
    const int wave = threadIdx.x >> 5;
    const int nw   = blockDim.x >> 5;
    for (int m = wave; m < Nv; m += nw) {
        float s = 0.f;
        for (int c = lane; c < C; c += 32) s += x[(long long)m * C + c];
        #pragma unroll
        for (int off = 16; off >= 1; off >>= 1) s += __shfl_xor(s, off, 32);
        if (lane == 0) lat[m] = s / (float)C;
    }
    __syncthreads();
    if ((int)threadIdx.x < S) {
        float acc = bout[threadIdx.x];
        for (int m = 0; m < Nv; ++m) acc += lat[m] * wout[threadIdx.x * Nv + m];
        dout[threadIdx.x] = 1.0f / (1.0f + expf(-acc));
    }
    if ((int)threadIdx.x < Nv) dout[S + threadIdx.x] = lat[threadIdx.x];
}

// ---------------------------------------------------------------------------
// Host
// ---------------------------------------------------------------------------
static inline int ilog2(int v) { int l = 0; while ((1 << l) < v) ++l; return l; }

extern "C" void kernel_launch(void* const* d_in, const int* in_sizes, int n_in,
                              void* d_out, int out_size, void* d_ws, size_t ws_size,
                              hipStream_t stream)
{
    static const int Ns[6]  = {40962, 10242, 2562, 642, 162, 42};
    static const int chs[7] = {2, 64, 128, 256, 512, 1024, 2048};

    const float* x_in = (const float*)d_in[0];
    const int* no[6];
    for (int i = 0; i < 6; ++i) no[i] = (const int*)d_in[1 + i];

    // params flattening: insertion order (wa,ba,ga,bea,wb,bb,gb,beb) vs
    // sorted-key order (ba,bb,bea,beb,ga,gb,wa,wb). Detect via size of entry 7:
    // insertion -> wa of block0 = 64*7*2 = 896; sorted -> ba = 64.
    const bool sortedKeys = (in_sizes[7] == chs[1]);
    const float *wa[6], *ba[6], *ga[6], *bea[6], *wb[6], *bb[6], *gb[6], *beb[6];
    for (int i = 0; i < 6; ++i) {
        void* const* p = d_in + 7 + 8 * i;
        if (!sortedKeys) {
            wa[i]  = (const float*)p[0]; ba[i]  = (const float*)p[1];
            ga[i]  = (const float*)p[2]; bea[i] = (const float*)p[3];
            wb[i]  = (const float*)p[4]; bb[i]  = (const float*)p[5];
            gb[i]  = (const float*)p[6]; beb[i] = (const float*)p[7];
        } else {
            ba[i]  = (const float*)p[0]; bb[i]  = (const float*)p[1];
            bea[i] = (const float*)p[2]; beb[i] = (const float*)p[3];
            ga[i]  = (const float*)p[4]; gb[i]  = (const float*)p[5];
            wa[i]  = (const float*)p[6]; wb[i]  = (const float*)p[7];
        }
    }
    const float* wout = (const float*)d_in[7 + 48];
    const float* bout = (const float*)d_in[7 + 49];

    // Ping-pong activation buffers (max activation: 40962*64 floats = 10.5 MB each)
    const size_t bufElems = (size_t)40962 * 64;
    float* A = (float*)d_ws;
    float* B = A + bufElems;

    const float* cur = x_in;
    for (int i = 0; i < 6; ++i) {
        const int Nv = Ns[i], cin = chs[i], cout = chs[i + 1];

        // convA: gather + GEMM + bias
        {
            const int tiles  = ((Nv + 15) / 16) * (cout / 16);
            const int blocks = (tiles + 7) / 8;
            gemm_gather_wmma<<<blocks, 256, 0, stream>>>(
                cur, no[i], wa[i], ba[i], A, Nv, cin, ilog2(cin), cout);
            gn_leaky<<<(Nv + 7) / 8, 256, 0, stream>>>(A, ga[i], bea[i], Nv, cout);
        }
        // convB
        {
            const int tiles  = ((Nv + 15) / 16) * (cout / 16);
            const int blocks = (tiles + 7) / 8;
            gemm_gather_wmma<<<blocks, 256, 0, stream>>>(
                A, no[i], wb[i], bb[i], B, Nv, cout, ilog2(cout), cout);
            gn_leaky<<<(Nv + 7) / 8, 256, 0, stream>>>(B, gb[i], beb[i], Nv, cout);
        }
        if (i < 5) {
            const int nc = Ns[i + 1];
            const long long total = (long long)nc * cout;
            pool_mean_kernel<<<(int)((total + 255) / 256), 256, 0, stream>>>(
                B, no[i], A, nc, cout);
            cur = A;
            float* t = A; A = B; B = t;   // swap roles for next level
        } else {
            cur = B;
        }
    }

    head_kernel<<<1, 256, 0, stream>>>(cur, wout, bout, (float*)d_out, 42, 2048, 4);
    (void)n_in; (void)out_size; (void)ws_size;
}
